// MoE_RQVAE_no_cf_48241072668752
// MI455X (gfx1250) — compile-verified, matrix operand order checked by reading, not run on hardware
//
#include <hip/hip_runtime.h>

typedef unsigned short u16;
typedef __attribute__((ext_vector_type(16))) __bf16 v16bf;
typedef __attribute__((ext_vector_type(8)))  float  v8f;

__device__ __forceinline__ u16 f2bf(float f) {
    unsigned int u = __float_as_uint(f);
    u = (u + 0x7FFFu + ((u >> 16) & 1u)) >> 16;
    return (u16)u;
}

// ---------------------------------------------------------------------------
// f32 -> bf16 conversion (n must be a multiple of 4; all our sizes are)
// ---------------------------------------------------------------------------
__global__ void cvt_bf16_kernel(const float* __restrict__ in, u16* __restrict__ out, size_t n) {
    size_t i = ((size_t)blockIdx.x * blockDim.x + threadIdx.x) * 4;
    if (i < n) {
        float4 v = *(const float4*)(in + i);
        unsigned long long p =  (unsigned long long)f2bf(v.x)
                             | ((unsigned long long)f2bf(v.y) << 16)
                             | ((unsigned long long)f2bf(v.z) << 32)
                             | ((unsigned long long)f2bf(v.w) << 48);
        *(unsigned long long*)(out + i) = p;
    }
}

__global__ void zero_kernel(float* p) { p[0] = 0.0f; }

__global__ void finalize_loss_kernel(const float* acc, float* out) {
    // loss = sum_levels (1+beta) * mean(diff^2) over (B*128) elements
    out[0] = acc[0] * (1.0f + 1.0e-3f) / (32768.0f * 128.0f);
}

// ---------------------------------------------------------------------------
// bf16 WMMA GEMM:  C[M,N] = act(A[M,K] @ W[K,N] + bias[N])
//   MODE 0: ReLU, store bf16 (intermediate layer)
//   MODE 1: no act, store f32 (encoder z / decoder output)
// Block tile 128x128, K-slab 64 (2 WMMA K-steps per stage), 8 waves (4x2),
// each wave computes 32x64 via 2x4 WMMA tiles -> 16 WMMAs between barriers.
// Software pipeline: fetch next slab to registers -> WMMA current -> stage to
// LDS -> barrier. No min-blocks hint: staging registers must stay resident
// (launch_bounds(256,2) caused scratch spills of aReg/bReg).
// ---------------------------------------------------------------------------
#define TM 128
#define TN 128
#define TK 64

union FragAB { uint4 q[2]; v16bf v; };

template <int MODE>
__global__ __launch_bounds__(256)
void gemm_bias_act_kernel(const u16* __restrict__ A, const u16* __restrict__ W,
                          const float* __restrict__ bias,
                          float* __restrict__ outF, u16* __restrict__ outB,
                          int N, int K) {
    __shared__ __align__(16) u16 As[2][TM * TK];   // 2 x 16 KB
    __shared__ __align__(16) u16 Bs[2][TN * TK];   // 2 x 16 KB

    const int t    = threadIdx.x;
    const int wave = t >> 5;
    const int lane = t & 31;
    const int wm   = wave >> 1;          // 0..3
    const int wn   = wave & 1;           // 0..1
    const int lh   = lane >> 4;          // lane half
    const int lr   = lane & 15;

    const int m0 = blockIdx.y * TM;
    const int n0 = blockIdx.x * TN;

    // per-thread staging assignments (constant across iterations)
    const int arow = t >> 1;             // A: one 64B contiguous run per thread
    const int akb  = (t & 1) * 32;
    const int bk2  = (t >> 4) * 2;       // B: two k-row pairs per thread
    const int bnc  = (t & 15) * 8;

    v8f acc[2][4];
#pragma unroll
    for (int mt = 0; mt < 2; ++mt)
#pragma unroll
        for (int nt = 0; nt < 4; ++nt)
            acc[mt][nt] = v8f{0.f, 0.f, 0.f, 0.f, 0.f, 0.f, 0.f, 0.f};

    uint4 aReg[4], bReg[4];

    auto fetch = [&](int k0) {           // global -> registers (loads only)
#pragma unroll
        for (int i = 0; i < 4; ++i)
            aReg[i] = *(const uint4*)(A + (size_t)(m0 + arow) * K + k0 + akb + i * 8);
        bReg[0] = *(const uint4*)(W + (size_t)(k0 + bk2)      * N + n0 + bnc);
        bReg[1] = *(const uint4*)(W + (size_t)(k0 + bk2 + 1)  * N + n0 + bnc);
        bReg[2] = *(const uint4*)(W + (size_t)(k0 + bk2 + 32) * N + n0 + bnc);
        bReg[3] = *(const uint4*)(W + (size_t)(k0 + bk2 + 33) * N + n0 + bnc);
    };
    auto stage = [&](int buf) {          // registers -> LDS (stores only)
#pragma unroll
        for (int i = 0; i < 4; ++i)
            *(uint4*)&As[buf][arow * TK + akb + i * 8] = aReg[i];
        const u16* p0 = (const u16*)&bReg[0];
        const u16* p1 = (const u16*)&bReg[1];
        const u16* p2 = (const u16*)&bReg[2];
        const u16* p3 = (const u16*)&bReg[3];
#pragma unroll
        for (int j = 0; j < 8; ++j) {    // transpose into [n][k], dword-packed
            unsigned int v0 = (unsigned int)p0[j] | ((unsigned int)p1[j] << 16);
            unsigned int v1 = (unsigned int)p2[j] | ((unsigned int)p3[j] << 16);
            *(unsigned int*)&Bs[buf][(bnc + j) * TK + bk2]      = v0;
            *(unsigned int*)&Bs[buf][(bnc + j) * TK + bk2 + 32] = v1;
        }
    };

    fetch(0);
    stage(0);
    __syncthreads();

    const int nk = K / TK;
    for (int kk = 0; kk < nk; ++kk) {
        const int cur = kk & 1;
        const bool more = (kk + 1 < nk);
        if (more) fetch((kk + 1) * TK);  // issue next-slab global loads early

#pragma unroll
        for (int h2 = 0; h2 < 2; ++h2) { // two WMMA K-steps per slab
            const int kb = 32 * h2;
            // A fragments: lane half holds K {0-7,16-23} / {8-15,24-31}
            FragAB af[2], bf[4];
#pragma unroll
            for (int mt = 0; mt < 2; ++mt) {
                int row = wm * 32 + mt * 16 + lr;
                af[mt].q[0] = *(const uint4*)&As[cur][row * TK + kb + 8 * lh];
                af[mt].q[1] = *(const uint4*)&As[cur][row * TK + kb + 16 + 8 * lh];
            }
#pragma unroll
            for (int nt = 0; nt < 4; ++nt) {
                int col = wn * 64 + nt * 16 + lr;
                bf[nt].q[0] = *(const uint4*)&Bs[cur][col * TK + kb + 8 * lh];
                bf[nt].q[1] = *(const uint4*)&Bs[cur][col * TK + kb + 16 + 8 * lh];
            }
#pragma unroll
            for (int mt = 0; mt < 2; ++mt)
#pragma unroll
                for (int nt = 0; nt < 4; ++nt)
                    acc[mt][nt] = __builtin_amdgcn_wmma_f32_16x16x32_bf16(
                        false, af[mt].v, false, bf[nt].v,
                        (short)0, acc[mt][nt], false, false);
        }

        if (more) stage(cur ^ 1);        // load-wait lands here, post-WMMA
        __syncthreads();
    }

    // Epilogue: bias + act + store. C layout: M = r + 8*lh, N = lr.
#pragma unroll
    for (int mt = 0; mt < 2; ++mt) {
#pragma unroll
        for (int nt = 0; nt < 4; ++nt) {
            int   col  = n0 + wn * 64 + nt * 16 + lr;
            float bval = bias[col];
#pragma unroll
            for (int r = 0; r < 8; ++r) {
                int   row = m0 + wm * 32 + mt * 16 + 8 * lh + r;
                float v   = acc[mt][nt][r] + bval;
                if (MODE == 0) {
                    v = fmaxf(v, 0.0f);
                    outB[(size_t)row * N + col] = f2bf(v);
                } else {
                    outF[(size_t)row * N + col] = v;
                }
            }
        }
    }
}

// ---------------------------------------------------------------------------
// Residual quantizer: one wave32 per sample, 8 samples per block.
// cb: [4, 10, 256, 128] f32. Writes indices (as float), x_q f32, x_q bf16,
// and atomically accumulates sum of (q - res)^2.
// ---------------------------------------------------------------------------
__global__ __launch_bounds__(256)
void rq_kernel(const float* __restrict__ z, const int* __restrict__ labels,
               const float* __restrict__ cb,
               float* __restrict__ out_idx, float* __restrict__ out_xq,
               u16* __restrict__ xqb, float* __restrict__ loss_acc) {
    __shared__ __align__(16) float res_s[8][128];

    const int t    = threadIdx.x;
    const int w    = t >> 5;
    const int lane = t & 31;
    const int s    = blockIdx.x * 8 + w;
    const int label = labels[s];

    float qacc[4] = {0.f, 0.f, 0.f, 0.f};
#pragma unroll
    for (int i = 0; i < 4; ++i)
        res_s[w][lane + 32 * i] = z[(size_t)s * 128 + lane + 32 * i];
    __syncthreads();

    float lsum = 0.0f;
    const float4* rs = (const float4*)res_s[w];

    for (int l = 0; l < 4; ++l) {
        const float* cbl = cb + ((size_t)(l * 10 + label) * 256) * 128;
        float best = 3.4e38f;
        int   bidx = 0;
#pragma unroll
        for (int j = 0; j < 8; ++j) {
            int k = lane + 32 * j;
            const float4* cr = (const float4*)(cbl + (size_t)k * 128);
            float dot = 0.f, cn = 0.f;
#pragma unroll 8
            for (int d = 0; d < 32; ++d) {
                float4 c = cr[d];
                float4 r = rs[d];
                dot += c.x * r.x + c.y * r.y + c.z * r.z + c.w * r.w;
                cn  += c.x * c.x + c.y * c.y + c.z * c.z + c.w * c.w;
            }
            float sc = cn - 2.0f * dot;     // ||res||^2 is constant per sample
            if (sc < best) { best = sc; bidx = k; }
        }
        // wave-wide argmin (tie -> lowest index, matching argmin semantics)
        for (int off = 16; off; off >>= 1) {
            float ob = __shfl_xor(best, off, 32);
            int   oi = __shfl_xor(bidx, off, 32);
            if (ob < best || (ob == best && oi < bidx)) { best = ob; bidx = oi; }
        }

        const float* q = cbl + (size_t)bidx * 128;
#pragma unroll
        for (int i = 0; i < 4; ++i) {
            int   d  = lane + 32 * i;
            float qv = q[d];
            float r  = res_s[w][d];
            float df = qv - r;
            lsum += df * df;
            qacc[i] += qv;
            res_s[w][d] = r - qv;           // residual update (in-wave LDS order)
        }
        if (lane == 0) out_idx[(size_t)s * 4 + l] = (float)bidx;
    }

    for (int off = 16; off; off >>= 1) lsum += __shfl_xor(lsum, off, 32);
    if (lane == 0) atomicAdd(loss_acc, lsum);

#pragma unroll
    for (int i = 0; i < 4; ++i) {
        int d = lane + 32 * i;
        out_xq[(size_t)s * 128 + d] = qacc[i];
        xqb[(size_t)s * 128 + d]    = f2bf(qacc[i]);
    }
}

// ---------------------------------------------------------------------------
// Host-side orchestration
// ---------------------------------------------------------------------------
static void run_cvt(const float* in, u16* out, size_t n, hipStream_t s) {
    int blocks = (int)((n / 4 + 255) / 256);
    cvt_bf16_kernel<<<blocks, 256, 0, s>>>(in, out, n);
}

extern "C" void kernel_launch(void* const* d_in, const int* in_sizes, int n_in,
                              void* d_out, int out_size, void* d_ws, size_t ws_size,
                              hipStream_t stream) {
    const float* x      = (const float*)d_in[0];
    const int*   labels = (const int*)d_in[1];
    const float* ew[4], *eb[4], *dw[4], *db[4];
    for (int l = 0; l < 4; ++l) { ew[l] = (const float*)d_in[2 + 2 * l];  eb[l] = (const float*)d_in[3 + 2 * l]; }
    for (int l = 0; l < 4; ++l) { dw[l] = (const float*)d_in[10 + 2 * l]; db[l] = (const float*)d_in[11 + 2 * l]; }
    const float* cb = (const float*)d_in[18];

    const int B = 32768;
    const int E[5] = {768, 2048, 1024, 512, 128};   // DEC dims are reversed

    // ---- workspace carve ----
    char* base = (char*)d_ws;
    size_t off = 0;
    auto carve = [&](size_t bytes) -> void* {
        off = (off + 255) & ~(size_t)255;
        void* p = base + off;
        off += bytes;
        return p;
    };
    u16 *ewb[4], *dwb[4];
    for (int l = 0; l < 4; ++l) ewb[l] = (u16*)carve((size_t)E[l] * E[l + 1] * 2);
    for (int l = 0; l < 4; ++l) dwb[l] = (u16*)carve((size_t)E[4 - l] * E[3 - l] * 2);
    u16*   xb    = (u16*)carve((size_t)B * 768 * 2);
    u16*   a2048 = (u16*)carve((size_t)B * 2048 * 2);   // h1, later d3
    u16*   a1024 = (u16*)carve((size_t)B * 1024 * 2);   // h2, later d2
    u16*   a512  = (u16*)carve((size_t)B * 512 * 2);    // h3, later d1
    float* zf    = (float*)carve((size_t)B * 128 * 4);
    u16*   xqb   = (u16*)carve((size_t)B * 128 * 2);
    float* lacc  = (float*)carve(4);

    // ---- d_out layout: out[B*768], loss[1], indices[B*4], x_q[B*128] ----
    float* outP     = (float*)d_out;
    float* out_dec  = outP;
    float* out_loss = outP + (size_t)B * 768;
    float* out_idx  = out_loss + 1;
    float* out_xq   = out_idx + (size_t)B * 4;

    // ---- conversions to bf16 ----
    run_cvt(x, xb, (size_t)B * 768, stream);
    for (int l = 0; l < 4; ++l) run_cvt(ew[l], ewb[l], (size_t)E[l] * E[l + 1], stream);
    for (int l = 0; l < 4; ++l) run_cvt(dw[l], dwb[l], (size_t)E[4 - l] * E[3 - l], stream);
    zero_kernel<<<1, 1, 0, stream>>>(lacc);

    auto gemm = [&](const u16* A, const u16* W, const float* bias,
                    float* oF, u16* oB, int N, int K, bool relu_bf16) {
        dim3 grid(N / 128, B / 128);
        if (relu_bf16)
            gemm_bias_act_kernel<0><<<grid, 256, 0, stream>>>(A, W, bias, nullptr, oB, N, K);
        else
            gemm_bias_act_kernel<1><<<grid, 256, 0, stream>>>(A, W, bias, oF, nullptr, N, K);
    };

    // ---- encoder ----
    gemm(xb,    ewb[0], eb[0], nullptr, a2048, 2048, 768,  true);
    gemm(a2048, ewb[1], eb[1], nullptr, a1024, 1024, 2048, true);
    gemm(a1024, ewb[2], eb[2], nullptr, a512,  512,  1024, true);
    gemm(a512,  ewb[3], eb[3], zf,      nullptr, 128, 512, false);   // z (no relu)

    // ---- residual quantization ----
    rq_kernel<<<B / 8, 256, 0, stream>>>(zf, labels, cb, out_idx, out_xq, xqb, lacc);
    finalize_loss_kernel<<<1, 1, 0, stream>>>(lacc, out_loss);

    // ---- decoder ----
    gemm(xqb,   dwb[0], db[0], nullptr, a512,  512,  128,  true);
    gemm(a512,  dwb[1], db[1], nullptr, a1024, 1024, 512,  true);
    gemm(a1024, dwb[2], db[2], nullptr, a2048, 2048, 1024, true);
    gemm(a2048, dwb[3], db[3], out_dec, nullptr, 768, 2048, false);  // output (no relu)
}
